// MLAAttention_41996190220293
// MI455X (gfx1250) — matrix operand order repaired and checked by reading, not compiled, not run on hardware
//
#include <hip/hip_runtime.h>
#include <hip/hip_bf16.h>
#include <math.h>

typedef __attribute__((ext_vector_type(16))) _Float16 v16h;
typedef __attribute__((ext_vector_type(8)))  _Float16 v8h;
typedef __attribute__((ext_vector_type(4)))  _Float16 v4h;
typedef __attribute__((ext_vector_type(8)))  float    v8f;
typedef __attribute__((ext_vector_type(4)))  float    v4f;

#define HIDN  2048
#define NH    20
#define QRK   768
#define KVRK  512
#define DNOPE 192
#define DROPE 64
#define DVAL  256
#define HDIM  256            // DNOPE + DROPE
#define SQ    2048
#define QROW  (NH*HDIM)      // 5120
#define KVROW (NH*(DNOPE+DVAL)) // 8960
#define ATTN_SCALE 0.0625f   // 1/sqrt(256)

static __device__ __forceinline__ v8f wmma_f16(v16h a, v16h b, v8f c) {
  return __builtin_amdgcn_wmma_f32_16x16x32_f16(false, a, false, b, (short)0, c,
                                                false, false);
}

// CDNA5 async global->LDS copy (16B per lane), tracked by ASYNCcnt.
// LDS byte address = low 32 bits of the generic pointer (flat LDS aperture).
static __device__ __forceinline__ void async_ld_b128(void* lds,
                                                     const void* gptr) {
  unsigned int loff = (unsigned int)(unsigned long long)lds;
  unsigned long long ga = (unsigned long long)gptr;
  asm volatile("global_load_async_to_lds_b128 %0, %1, off"
               :: "v"(loff), "v"(ga)
               : "memory");
}
static __device__ __forceinline__ void async_wait0() {
  asm volatile("s_wait_asynccnt 0" ::: "memory");
}

// ---------------------------------------------------------------- f32 -> f16
__global__ __launch_bounds__(256) void cvt_f32_f16(const float* __restrict__ s,
                                                   _Float16* __restrict__ d,
                                                   int n4) {
  int i = blockIdx.x * 256 + threadIdx.x;
  if (i >= n4) return;
  v4f v = *(const v4f*)(s + (size_t)i * 4);
  v4h o;
  o[0] = (_Float16)v[0]; o[1] = (_Float16)v[1];
  o[2] = (_Float16)v[2]; o[3] = (_Float16)v[3];
  *(v4h*)(d + (size_t)i * 4) = o;
}

// ---------------------------------------------------- generic WMMA GEMM
// C(MxN) = A(MxK) * W(NxK)^T ; f16 inputs, f32 accumulate, f16 or f32 out.
// Block: 256 threads = 8 waves (4x2). Wave tile: 32 x (NF*16 * 2 cols / wave
// pair) -> block tile BM x (NF*32). Double-buffered async LDS staging.
#define BM 128
#define BK 32
#define LPAD 40

template <int NF>  // N-fragments per wave (wave tile = 32 x NF*16)
__global__ __launch_bounds__(256) void gemm_wmma(
    const _Float16* __restrict__ A, const _Float16* __restrict__ W,
    float* __restrict__ Cf, _Float16* __restrict__ Ch,
    int M, int N, int K, int out16) {
  constexpr int BNt = NF * 32;  // block N tile (2 waves across N)
  __shared__ alignas(16) _Float16 As[2][BM][LPAD];
  __shared__ alignas(16) _Float16 Ws[2][BNt][LPAD];
  const int tid = threadIdx.x;
  const int wid = tid >> 5, lane = tid & 31;
  const int l16 = lane & 15, hi = lane >> 4;
  const int wm = wid >> 1, wn = wid & 1;
  const int bm0 = blockIdx.y * BM, bn0 = blockIdx.x * BNt;
  const int kOff = hi * 8;    // A fragment: K in {kOff..kOff+7, 16+kOff..}
  const int bOff = hi * 8;    // B fragment: K in {bOff..bOff+7, 16+bOff..}

  auto stage = [&](int buf, int k0) {
#pragma unroll
    for (int c = tid; c < BM * 4; c += 256) {
      int r = c >> 2, cc = (c & 3) * 8;
      async_ld_b128(&As[buf][r][cc], A + (size_t)(bm0 + r) * K + k0 + cc);
    }
#pragma unroll
    for (int c = tid; c < BNt * 4; c += 256) {
      int r = c >> 2, cc = (c & 3) * 8;
      async_ld_b128(&Ws[buf][r][cc], W + (size_t)(bn0 + r) * K + k0 + cc);
    }
  };

  v8f acc[2][NF];
#pragma unroll
  for (int i = 0; i < 2; ++i)
#pragma unroll
    for (int j = 0; j < NF; ++j)
#pragma unroll
      for (int r = 0; r < 8; ++r) acc[i][j][r] = 0.f;

  stage(0, 0);
  int cur = 0;
  for (int k0 = 0; k0 < K; k0 += BK) {
    async_wait0();       // my copies for buffer `cur` are done
    __syncthreads();     // everyone's copies done; everyone done reading cur^1
    if (k0 + BK < K) stage(cur ^ 1, k0 + BK);  // overlap next copy w/ compute

    v16h bfr[NF];
#pragma unroll
    for (int ni = 0; ni < NF; ++ni) {
      int n = wn * (BNt / 2) + ni * 16 + l16;
      ((v8h*)&bfr[ni])[0] = *(const v8h*)&Ws[cur][n][bOff];
      ((v8h*)&bfr[ni])[1] = *(const v8h*)&Ws[cur][n][16 + bOff];
    }
#pragma unroll
    for (int mi = 0; mi < 2; ++mi) {
      int m = wm * 32 + mi * 16 + l16;
      v16h a;
      ((v8h*)&a)[0] = *(const v8h*)&As[cur][m][kOff];
      ((v8h*)&a)[1] = *(const v8h*)&As[cur][m][16 + kOff];
#pragma unroll
      for (int ni = 0; ni < NF; ++ni)
        acc[mi][ni] = wmma_f16(a, bfr[ni], acc[mi][ni]);
    }
    cur ^= 1;
  }

#pragma unroll
  for (int mi = 0; mi < 2; ++mi)
#pragma unroll
    for (int ni = 0; ni < NF; ++ni) {
      int mb = bm0 + wm * 32 + mi * 16 + hi * 8;
      int n  = bn0 + wn * (BNt / 2) + ni * 16 + l16;
      if (out16) {
#pragma unroll
        for (int r = 0; r < 8; ++r)
          Ch[(size_t)(mb + r) * N + n] = (_Float16)acc[mi][ni][r];
      } else {
#pragma unroll
        for (int r = 0; r < 8; ++r)
          Cf[(size_t)(mb + r) * N + n] = acc[mi][ni][r];
      }
    }
}

// ---------------------------------------------------- RMSNorm (q path)
__global__ __launch_bounds__(256) void q_rmsnorm(const float* __restrict__ src,
                                                 const float* __restrict__ w,
                                                 _Float16* __restrict__ dst) {
  __shared__ float red[8];
  const int s = blockIdx.x;
  const float* row = src + (size_t)s * QRK;
  float ss = 0.f;
  for (int i = threadIdx.x; i < QRK; i += 256) { float v = row[i]; ss += v * v; }
#pragma unroll
  for (int off = 16; off > 0; off >>= 1) ss += __shfl_xor(ss, off, 32);
  if ((threadIdx.x & 31) == 0) red[threadIdx.x >> 5] = ss;
  __syncthreads();
  float tot = 0.f;
#pragma unroll
  for (int i = 0; i < 8; ++i) tot += red[i];
  float inv = rsqrtf(tot / (float)QRK + 1e-5f);
  for (int i = threadIdx.x; i < QRK; i += 256)
    dst[(size_t)s * QRK + i] = (_Float16)(row[i] * inv * w[i]);
}

// ------------------------------------- kv split: RMSNorm(512) + RoPE(k_pe)
__global__ __launch_bounds__(256) void kv_norm_rope(
    const float* __restrict__ src, const float* __restrict__ w,
    const int* __restrict__ pos, _Float16* __restrict__ kvn,
    _Float16* __restrict__ kr) {
  __shared__ float red[8];
  const int s = blockIdx.x;
  const float* row = src + (size_t)s * (KVRK + DROPE);
  float ss = 0.f;
  for (int i = threadIdx.x; i < KVRK; i += 256) { float v = row[i]; ss += v * v; }
#pragma unroll
  for (int off = 16; off > 0; off >>= 1) ss += __shfl_xor(ss, off, 32);
  if ((threadIdx.x & 31) == 0) red[threadIdx.x >> 5] = ss;
  __syncthreads();
  float tot = 0.f;
#pragma unroll
  for (int i = 0; i < 8; ++i) tot += red[i];
  float inv = rsqrtf(tot / (float)KVRK + 1e-5f);
  for (int i = threadIdx.x; i < KVRK; i += 256)
    kvn[(size_t)s * KVRK + i] = (_Float16)(row[i] * inv * w[i]);
  if (threadIdx.x < DROPE / 2) {
    int i = threadIdx.x;
    float a = row[KVRK + 2 * i], b = row[KVRK + 2 * i + 1];
    float fr = (float)pos[s] * __powf(1.0e6f, -(float)(2 * i) / (float)DROPE);
    float cs = __cosf(fr), sn = __sinf(fr);
    kr[(size_t)s * DROPE + i]              = (_Float16)(a * cs - b * sn);
    kr[(size_t)s * DROPE + DROPE / 2 + i]  = (_Float16)(a * sn + b * cs);
  }
}

// ------------------------------------- RoPE on q (last 64 dims of each head)
__global__ __launch_bounds__(640) void q_rope(_Float16* __restrict__ q,
                                              const int* __restrict__ pos) {
  const int s = blockIdx.x;
  const int h = threadIdx.x >> 5;
  const int i = threadIdx.x & 31;
  _Float16* base = q + (size_t)s * QROW + h * HDIM + DNOPE;
  float a = (float)base[2 * i], b = (float)base[2 * i + 1];
  float fr = (float)pos[s] * __powf(1.0e6f, -(float)(2 * i) / (float)DROPE);
  float cs = __cosf(fr), sn = __sinf(fr);
  __syncthreads();  // all reads complete before in-place writes
  base[i]      = (_Float16)(a * cs - b * sn);
  base[32 + i] = (_Float16)(a * sn + b * cs);
}

// ---------------------------------------------------- flash attention (MLA)
// block: 8 waves x 16 q-rows = 128 q rows, one head per blockIdx.y
#define FKT 32
#define KPAD 40

__global__ __launch_bounds__(256) void mla_fa(
    const _Float16* __restrict__ Q, const _Float16* __restrict__ KVX,
    const _Float16* __restrict__ KR, _Float16* __restrict__ Out) {
  __shared__ alignas(16) _Float16 Kt[FKT][HDIM];   // 32 keys x 256 (nope||rope)
  __shared__ alignas(16) _Float16 Vt[DVAL][KPAD];  // V^T: dv-major, key contig
  __shared__ alignas(16) _Float16 Ps[8][16][KPAD]; // per-wave prob staging
  const int tid = threadIdx.x;
  const int wid = tid >> 5, lane = tid & 31;
  const int l16 = lane & 15, hi = lane >> 4;
  const int h = blockIdx.y;
  const int q0 = blockIdx.x * 128 + wid * 16;
  const int kOff = hi * 8;
  const int bOff = hi * 8;

  // resident Q fragments: 8 chunks of K=32 across head dim 256
  v16h qf[8];
  {
    const _Float16* qrow = Q + (size_t)(q0 + l16) * QROW + h * HDIM;
#pragma unroll
    for (int c = 0; c < 8; ++c) {
      ((v8h*)&qf[c])[0] = *(const v8h*)(qrow + c * 32 + kOff);
      ((v8h*)&qf[c])[1] = *(const v8h*)(qrow + c * 32 + 16 + kOff);
    }
  }

  v8f oacc[16];
#pragma unroll
  for (int c = 0; c < 16; ++c)
#pragma unroll
    for (int r = 0; r < 8; ++r) oacc[c][r] = 0.f;
  float rm[8], rl[8];
#pragma unroll
  for (int r = 0; r < 8; ++r) { rm[r] = -1e30f; rl[r] = 0.f; }

  for (int kb = 0; kb < SQ; kb += FKT) {
    // async-stage K rows (gather nope || rope); overlaps with V transpose below
#pragma unroll
    for (int c = tid; c < FKT * 32; c += 256) {
      int r = c >> 5, d = (c & 31) * 8;
      const _Float16* src =
          (d < DNOPE)
              ? (KVX + (size_t)(kb + r) * KVROW + h * (DNOPE + DVAL) + d)
              : (KR + (size_t)(kb + r) * DROPE + (d - DNOPE));
      async_ld_b128(&Kt[r][d], src);
    }
    // stage V transposed (needs register transpose, so load+ds_store path)
    for (int c = tid; c < FKT * 32; c += 256) {
      int r = c >> 5, d = (c & 31) * 8;
      v8h vv = *(const v8h*)(KVX + (size_t)(kb + r) * KVROW +
                             h * (DNOPE + DVAL) + DNOPE + d);
#pragma unroll
      for (int i = 0; i < 8; ++i) Vt[d + i][r] = vv[i];
    }
    async_wait0();
    __syncthreads();

    // scores: 16 q rows x 32 keys
    v8f s0, s1;
#pragma unroll
    for (int r = 0; r < 8; ++r) { s0[r] = 0.f; s1[r] = 0.f; }
#pragma unroll
    for (int c = 0; c < 8; ++c) {
      v16h b0, b1;
      ((v8h*)&b0)[0] = *(const v8h*)&Kt[l16][c * 32 + bOff];
      ((v8h*)&b0)[1] = *(const v8h*)&Kt[l16][c * 32 + 16 + bOff];
      ((v8h*)&b1)[0] = *(const v8h*)&Kt[16 + l16][c * 32 + bOff];
      ((v8h*)&b1)[1] = *(const v8h*)&Kt[16 + l16][c * 32 + 16 + bOff];
      s0 = wmma_f16(qf[c], b0, s0);
      s1 = wmma_f16(qf[c], b1, s1);
    }

    // online softmax; C-layout row m = 8*hi + r lives across one 16-lane half
    float pscale[8];
#pragma unroll
    for (int r = 0; r < 8; ++r) {
      float e0 = s0[r] * ATTN_SCALE;
      float e1 = s1[r] * ATTN_SCALE;
      float t = fmaxf(e0, e1);
#pragma unroll
      for (int off = 8; off > 0; off >>= 1) t = fmaxf(t, __shfl_xor(t, off, 16));
      float mnew = fmaxf(rm[r], t);
      pscale[r] = __expf(rm[r] - mnew);
      rm[r] = mnew;
      float p0 = __expf(e0 - mnew);
      float p1 = __expf(e1 - mnew);
      Ps[wid][hi * 8 + r][l16]      = (_Float16)p0;
      Ps[wid][hi * 8 + r][l16 + 16] = (_Float16)p1;
      float rs = p0 + p1;
#pragma unroll
      for (int off = 8; off > 0; off >>= 1) rs += __shfl_xor(rs, off, 16);
      rl[r] = rl[r] * pscale[r] + rs;
    }
    asm volatile("s_wait_dscnt 0" ::: "memory");  // P staging visible (same wave)

    v16h pf;  // probabilities as A fragment
    ((v8h*)&pf)[0] = *(const v8h*)&Ps[wid][l16][kOff];
    ((v8h*)&pf)[1] = *(const v8h*)&Ps[wid][l16][16 + kOff];

#pragma unroll
    for (int c = 0; c < 16; ++c) {
#pragma unroll
      for (int r = 0; r < 8; ++r) oacc[c][r] *= pscale[r];
      v16h b;
      ((v8h*)&b)[0] = *(const v8h*)&Vt[c * 16 + l16][bOff];
      ((v8h*)&b)[1] = *(const v8h*)&Vt[c * 16 + l16][16 + bOff];
      oacc[c] = wmma_f16(pf, b, oacc[c]);
    }
    __syncthreads();
  }

#pragma unroll
  for (int c = 0; c < 16; ++c)
#pragma unroll
    for (int r = 0; r < 8; ++r) {
      int m = q0 + hi * 8 + r;
      Out[(size_t)m * QROW + h * DVAL + c * 16 + l16] =
          (_Float16)(oacc[c][r] / rl[r]);
    }
}

// ---------------------------------------------------------------- launcher
extern "C" void kernel_launch(void* const* d_in, const int* in_sizes, int n_in,
                              void* d_out, int out_size, void* d_ws,
                              size_t ws_size, hipStream_t stream) {
  (void)in_sizes; (void)n_in; (void)out_size; (void)ws_size;
  const float* x    = (const float*)d_in[0];
  const int*   pos  = (const int*)d_in[1];
  const float* qaw  = (const float*)d_in[2];
  const float* qbw  = (const float*)d_in[3];
  const float* kvaw = (const float*)d_in[4];
  const float* kvbw = (const float*)d_in[5];
  const float* ow   = (const float*)d_in[6];
  const float* qln  = (const float*)d_in[7];
  const float* kvln = (const float*)d_in[8];

  char* p = (char*)d_ws;
  auto alloc = [&](size_t bytes) -> char* {
    char* r = p;
    p += (bytes + 255) & ~(size_t)255;
    return r;
  };
  _Float16* xh    = (_Float16*)alloc((size_t)SQ * HIDN * 2);
  _Float16* qawh  = (_Float16*)alloc((size_t)QRK * HIDN * 2);
  _Float16* qbwh  = (_Float16*)alloc((size_t)QROW * QRK * 2);
  _Float16* kvawh = (_Float16*)alloc((size_t)(KVRK + DROPE) * HIDN * 2);
  _Float16* kvbwh = (_Float16*)alloc((size_t)KVROW * KVRK * 2);
  _Float16* owh   = (_Float16*)alloc((size_t)HIDN * QROW * 2);
  float*    qc    = (float*)alloc((size_t)SQ * QRK * 4);
  float*    kvc   = (float*)alloc((size_t)SQ * (KVRK + DROPE) * 4);
  _Float16* qcn   = (_Float16*)alloc((size_t)SQ * QRK * 2);
  _Float16* kvn   = (_Float16*)alloc((size_t)SQ * KVRK * 2);
  _Float16* kr    = (_Float16*)alloc((size_t)SQ * DROPE * 2);
  _Float16* qh    = (_Float16*)alloc((size_t)SQ * QROW * 2);
  _Float16* kvexp = (_Float16*)alloc((size_t)SQ * KVROW * 2);
  _Float16* ao    = (_Float16*)alloc((size_t)SQ * QROW * 2);

  auto cvt = [&](const float* s, _Float16* d, size_t n) {
    int n4 = (int)(n / 4);
    cvt_f32_f16<<<(n4 + 255) / 256, 256, 0, stream>>>(s, d, n4);
  };
  cvt(x, xh, (size_t)SQ * HIDN);
  cvt(qaw, qawh, (size_t)QRK * HIDN);
  cvt(qbw, qbwh, (size_t)QROW * QRK);
  cvt(kvaw, kvawh, (size_t)(KVRK + DROPE) * HIDN);
  cvt(kvbw, kvbwh, (size_t)KVROW * KVRK);
  cvt(ow, owh, (size_t)HIDN * QROW);

  // q_c = x @ q_a_w^T (N=768 -> 128-wide tiles) ;
  // kv_c = x @ kv_a_w^T (N=576 -> 64-wide tiles)
  gemm_wmma<4><<<dim3(QRK / 128, SQ / BM), 256, 0, stream>>>(
      xh, qawh, qc, nullptr, SQ, QRK, HIDN, 0);
  gemm_wmma<2><<<dim3((KVRK + DROPE) / 64, SQ / BM), 256, 0, stream>>>(
      xh, kvawh, kvc, nullptr, SQ, KVRK + DROPE, HIDN, 0);

  q_rmsnorm<<<SQ, 256, 0, stream>>>(qc, qln, qcn);
  kv_norm_rope<<<SQ, 256, 0, stream>>>(kvc, kvln, pos, kvn, kr);

  // q = norm(q_c) @ q_b_w^T ; kv_exp = norm(kv_lora) @ kv_b_w^T
  gemm_wmma<4><<<dim3(QROW / 128, SQ / BM), 256, 0, stream>>>(
      qcn, qbwh, nullptr, qh, SQ, QROW, QRK, 1);
  q_rope<<<SQ, NH * 32, 0, stream>>>(qh, pos);
  gemm_wmma<4><<<dim3(KVROW / 128, SQ / BM), 256, 0, stream>>>(
      kvn, kvbwh, nullptr, kvexp, SQ, KVROW, KVRK, 1);

  // flash attention per head
  mla_fa<<<dim3(SQ / 128, NH), 256, 0, stream>>>(qh, kvexp, kr, ao);

  // out = attn_out @ o_w^T  (f32 to d_out)
  gemm_wmma<4><<<dim3(HIDN / 128, SQ / BM), 256, 0, stream>>>(
      ao, owh, (float*)d_out, nullptr, SQ, HIDN, QROW, 0);
}
